// FragmentAwareEncoder_30477087933033
// MI455X (gfx1250) — matrix operand order repaired and checked by reading, not compiled, not run on hardware
//
#include <hip/hip_runtime.h>
#include <hip/hip_bf16.h>

typedef __bf16 bf16;
typedef __attribute__((ext_vector_type(16))) __bf16 v16bf;
typedef __attribute__((ext_vector_type(8)))  float  v8f;
typedef __attribute__((ext_vector_type(4)))  float  v4f;

#define N_NODES 32768
#define HDIM    256
#define NGRAPH  512
#define EPG     512
#define NPG     64
#define NFRAG   16

// ---------------------------------------------------------------------------
// WMMA helpers (CDNA5 wave32, 16x16x32 bf16 -> f32)
// ---------------------------------------------------------------------------
__device__ __forceinline__ v8f v8f_zero() {
    v8f z = {0.f,0.f,0.f,0.f,0.f,0.f,0.f,0.f};
    return z;
}

__device__ __forceinline__ v8f wmma_bf16(v16bf a, v16bf b, v8f c) {
    // (neg_a, A, neg_b, B, c_mod, C, reuse_a, reuse_b)
    return __builtin_amdgcn_wmma_f32_16x16x32_bf16(false, a, false, b, (short)0, c, false, false);
}

union AFragU { v16bf v; v4f q[2]; };

// A (16x32 bf16) fragment from row-major [N x K] activations.
// Contiguous k order (packw matches): lanes 0-15 (M=lane) hold K = {0..15}; lanes 16-31 hold
// K = {16..31}.
__device__ __forceinline__ v16bf load_afrag(const bf16* __restrict__ A, int rowBase,
                                            int K, int kt, int lane) {
    int m    = rowBase + (lane & 15);
    int half = lane >> 4;
    const bf16* base = A + (size_t)m * K + kt * 32 + half * 16;
    AFragU u;
    u.q[0] = *(const v4f*)(base);
    u.q[1] = *(const v4f*)(base + 8);
    return u.v;
}

// CDNA5 async global->LDS copy (16 bytes/thread), tracked by ASYNCcnt.
// The LDS destination is passed as a REAL pointer into the __shared__ array:
// the ptrtoint both captures the allocation (so the "memory"-clobbering asm is
// assumed to write it -> later ds_loads are not folded to undef) and yields the
// LDS byte offset in its low 32 bits (ISA 10.2).
__device__ __forceinline__ void stage_async16(const bf16* gsrc, const bf16* lds_dst) {
    unsigned off32 = (unsigned)(unsigned long long)(const void*)lds_dst;
    unsigned long long ga = (unsigned long long)(const void*)gsrc;
    asm volatile("global_load_async_to_lds_b128 %0, %1, off"
                 :: "v"(off32), "v"(ga) : "memory");
}
__device__ __forceinline__ void wait_asynccnt0() {
    asm volatile("s_wait_asynccnt 0x0" ::: "memory");
}

// ---------------------------------------------------------------------------
// Pack kernels
// ---------------------------------------------------------------------------
__global__ void packx_kernel(const float* __restrict__ x, bf16* __restrict__ A0, int N) {
    int idx = blockIdx.x * 256 + threadIdx.x;
    if (idx >= N * 32) return;
    int m = idx >> 5, k = idx & 31;
    A0[idx] = (bf16)((k < 25) ? x[m * 25 + k] : 0.f);
}

// Pack W [Ksrc x 256] (row-major) into WMMA B-fragment order, K padded to Kpad.
// Bp index = kt*8192 + nt*512 + lane*16 + j ; n = nt*16+(lane&15), k = kt*32+(lane>>4)*16+j
__global__ void packw_kernel(const float* __restrict__ W, bf16* __restrict__ Bp,
                             int Kpad, int Ksrc) {
    int idx = blockIdx.x * 256 + threadIdx.x;
    int total = Kpad * HDIM;
    if (idx >= total) return;
    int j    = idx & 15;
    int lane = (idx >> 4) & 31;
    int nt   = (idx >> 9) & 15;
    int kt   = idx >> 13;
    int n = nt * 16 + (lane & 15);
    int k = kt * 32 + (lane >> 4) * 16 + j;
    float v = (k < Ksrc) ? W[k * HDIM + n] : 0.f;
    Bp[idx] = (bf16)v;
}

__global__ void fragid_kernel(const float* __restrict__ s, int* __restrict__ fid, int N) {
    int i = blockIdx.x * 256 + threadIdx.x;
    if (i >= N) return;
    int f = 0;
    float best = s[(size_t)i * NFRAG];
    for (int k = 1; k < NFRAG; ++k) {
        float v = s[(size_t)i * NFRAG + k];
        if (v > best) { best = v; f = k; }
    }
    fid[i] = f;
}

// ---------------------------------------------------------------------------
// Triple GEMM: C{0,1,2}[N x 256] = A[N x K] @ W{0,1,2} + bias{0,1,2}
// One wave: 16 rows x 64 cols for all three outputs, sharing the A fragment.
// B fragments are staged to LDS with double-buffered async copies; the 12
// WMMAs per kt-step consume them through a distance-2 register pipeline so
// ds_load latency is hidden behind ~2 WMMAs of math.
// Block = 256 threads = 8 waves -> 128 rows. grid = (N/128, 256/64).
// ---------------------------------------------------------------------------
#define BSTAGE_ELEMS 2048                 /* 4 KB per matrix chunk, bf16 elems */
#define BBUF_HALF    (3 * BSTAGE_ELEMS)   /* 6144 elems = 12 KB */

__global__ __launch_bounds__(256) void gemm3_kernel(
    const bf16* __restrict__ A, int K,
    const bf16* __restrict__ B0, const bf16* __restrict__ B1, const bf16* __restrict__ B2,
    const float* __restrict__ bias0, const float* __restrict__ bias1, const float* __restrict__ bias2,
    float* __restrict__ C0, float* __restrict__ C1, float* __restrict__ C2)
{
    __shared__ __align__(64) bf16 Bbuf[2 * BBUF_HALF];   // 24 KB double buffer

    const int tid     = threadIdx.x;
    const int lane    = tid & 31;
    const int wave    = tid >> 5;
    const int rowBase = blockIdx.x * 128 + wave * 16;
    const int colBase = blockIdx.y * 64;
    const int ntBase  = colBase >> 4;
    const int KT      = K >> 5;

    const bf16* Bmat[3] = { B0, B1, B2 };

    // stage(kt) -> Bbuf[buf]: each thread moves 16B per matrix (256 thr x 16B = 4KB chunk)
    auto stage = [&](int kt, int buf) {
#pragma unroll
        for (int i = 0; i < 3; ++i) {
            const bf16* src = Bmat[i] + ((size_t)(kt * 16 + ntBase) * 512) + tid * 8;
            const bf16* dst = Bbuf + (buf * BBUF_HALF + i * BSTAGE_ELEMS + tid * 8);
            stage_async16(src, dst);
        }
    };

    v8f acc0[4], acc1[4], acc2[4];
#pragma unroll
    for (int t = 0; t < 4; ++t) { acc0[t] = v8f_zero(); acc1[t] = v8f_zero(); acc2[t] = v8f_zero(); }

    stage(0, 0);
    v16bf a = load_afrag(A, rowBase, K, 0, lane);

    for (int kt = 0; kt < KT; ++kt) {
        const int cur = kt & 1;
        // my async copies into Bbuf[cur] done; barrier publishes them (and fences
        // the previous iteration's ds_loads of Bbuf[1-cur]).
        wait_asynccnt0();
        __syncthreads();
        if (kt + 1 < KT) stage(kt + 1, cur ^ 1);              // overlaps with WMMAs below
        int ktn = (kt + 1 < KT) ? kt + 1 : kt;
        v16bf a_next = load_afrag(A, rowBase, K, ktn, lane);  // overlaps too

        // Distance-2 pipelined consume: fragment q at LDS offset
        // (q>>2)*BSTAGE_ELEMS + (q&3)*512 ; keep 2 fragments in flight.
        const bf16* lb = Bbuf + cur * BBUF_HALF + lane * 16;
        v16bf b0 = *(const v16bf*)(lb);
        v16bf b1 = *(const v16bf*)(lb + 512);
#pragma unroll
        for (int q = 0; q < 12; ++q) {
            int qn = (q + 2 < 12) ? q + 2 : 11;
            v16bf bn = *(const v16bf*)(lb + (qn >> 2) * BSTAGE_ELEMS + (qn & 3) * 512);
            if (q < 4)       acc0[q]     = wmma_bf16(a, b0, acc0[q]);
            else if (q < 8)  acc1[q - 4] = wmma_bf16(a, b0, acc1[q - 4]);
            else             acc2[q - 8] = wmma_bf16(a, b0, acc2[q - 8]);
            b0 = b1;
            b1 = bn;
        }
        a = a_next;
    }

    const int half = lane >> 4;
    const int nlow = lane & 15;
#pragma unroll
    for (int t = 0; t < 4; ++t) {
        int n = colBase + t * 16 + nlow;
        float bs0 = bias0[n], bs1 = bias1[n], bs2 = bias2[n];
#pragma unroll
        for (int r = 0; r < 8; ++r) {
            size_t off = (size_t)(rowBase + r + 8 * half) * HDIM + n;
            C0[off] = acc0[t][r] + bs0;
            C1[off] = acc1[t][r] + bs1;
            C2[off] = acc2[t][r] + bs2;
        }
    }
}

// ---------------------------------------------------------------------------
// Edge aggregation + residual self term + layernorm + ReLU. One block = one
// graph; thread t owns column t so edge scatter into LDS needs no atomics.
// Edges processed in chunks of 8 so the gather loads overlap.
// ---------------------------------------------------------------------------
__global__ __launch_bounds__(256) void fuse_kernel(
    const float* __restrict__ Hs, const float* __restrict__ Ha, const float* __restrict__ Hr,
    const int* __restrict__ esrc, const int* __restrict__ edst,
    const unsigned char* __restrict__ emask,
    const float* __restrict__ gamma, const float* __restrict__ beta,
    bf16* __restrict__ Aout, float* __restrict__ Fout)
{
    __shared__ float agg[NPG * HDIM];          // exactly 64 KB
    const int g   = blockIdx.x;
    const int tid = threadIdx.x;

    for (int i = tid; i < NPG * HDIM; i += 256) agg[i] = 0.f;
    __syncthreads();

    const int c = tid;
    const int ebase = g * EPG;
    for (int e0 = 0; e0 < EPG; e0 += 8) {
        float vals[8];
        int   dloc[8];
#pragma unroll
        for (int q = 0; q < 8; ++q) {
            int e   = ebase + e0 + q;
            int src = esrc[e];
            dloc[q] = edst[e] & (NPG - 1);
            const float* hrow = ((emask[e] != 0) ? Ha : Hr) + (size_t)src * HDIM;
            vals[q] = hrow[c];
        }
#pragma unroll
        for (int q = 0; q < 8; ++q)
            agg[dloc[q] * HDIM + c] += vals[q];
    }
    __syncthreads();

    const int wave = tid >> 5;
    const int lane = tid & 31;
    for (int r = wave; r < NPG; r += 8) {
        int node = g * NPG + r;
        float v[8];
        float sum = 0.f, sumsq = 0.f;
#pragma unroll
        for (int i = 0; i < 8; ++i) {
            int cc = i * 32 + lane;
            float t = Hs[(size_t)node * HDIM + cc] + agg[r * HDIM + cc];
            v[i] = t; sum += t; sumsq += t * t;
        }
        for (int off = 16; off > 0; off >>= 1) {
            sum   += __shfl_xor(sum,   off, 32);
            sumsq += __shfl_xor(sumsq, off, 32);
        }
        float mu  = sum * (1.f / HDIM);
        float var = sumsq * (1.f / HDIM) - mu * mu;
        float inv = rsqrtf(var + 1e-5f);
#pragma unroll
        for (int i = 0; i < 8; ++i) {
            int cc = i * 32 + lane;
            float y = (v[i] - mu) * inv * gamma[cc] + beta[cc];
            y = fmaxf(y, 0.f);
            if (Aout) Aout[(size_t)node * HDIM + cc] = (bf16)y;
            if (Fout) Fout[(size_t)node * HDIM + cc] = y;
        }
    }
}

// ---------------------------------------------------------------------------
// Fragment pooling (one-hot segment sum) + layernorm + fragment mask.
// ---------------------------------------------------------------------------
__global__ __launch_bounds__(256) void frag_kernel(
    const float* __restrict__ H, const int* __restrict__ fid,
    const float* __restrict__ fgamma, const float* __restrict__ fbeta,
    float* __restrict__ frag, float* __restrict__ fmask)
{
    __shared__ float acc[NFRAG * HDIM];
    __shared__ int cnt[NFRAG];
    const int g = blockIdx.x, tid = threadIdx.x, c = tid;

    for (int i = tid; i < NFRAG * HDIM; i += 256) acc[i] = 0.f;
    if (tid < NFRAG) cnt[tid] = 0;
    __syncthreads();

    for (int n = 0; n < NPG; ++n) {
        int node = g * NPG + n;
        int f = fid[node];
        acc[f * HDIM + c] += H[(size_t)node * HDIM + c];
        if (tid == 0) cnt[f]++;
    }
    __syncthreads();

    const int wave = tid >> 5, lane = tid & 31;
    for (int r = wave; r < NFRAG; r += 8) {
        float v[8];
        float sum = 0.f, sumsq = 0.f;
#pragma unroll
        for (int i = 0; i < 8; ++i) {
            int cc = i * 32 + lane;
            float t = acc[r * HDIM + cc];
            v[i] = t; sum += t; sumsq += t * t;
        }
        for (int off = 16; off > 0; off >>= 1) {
            sum   += __shfl_xor(sum,   off, 32);
            sumsq += __shfl_xor(sumsq, off, 32);
        }
        float mu  = sum * (1.f / HDIM);
        float var = sumsq * (1.f / HDIM) - mu * mu;
        float inv = rsqrtf(var + 1e-5f);
#pragma unroll
        for (int i = 0; i < 8; ++i) {
            int cc = i * 32 + lane;
            frag[((size_t)g * NFRAG + r) * HDIM + cc] = (v[i] - mu) * inv * fgamma[cc] + fbeta[cc];
        }
    }
    if (tid < NFRAG) fmask[g * NFRAG + tid] = (cnt[tid] > 0) ? 1.f : 0.f;
}

// ---------------------------------------------------------------------------
// reg_loss = sum|W_inter0| + sum|b_inter0| + sum|W_inter| + sum|b_inter|
// ---------------------------------------------------------------------------
__global__ __launch_bounds__(256) void regloss_kernel(
    const float* __restrict__ Wi0, const float* __restrict__ bi0,
    const float* __restrict__ Wi,  const float* __restrict__ bi,
    float* __restrict__ out)
{
    __shared__ float red[256];
    const int tid = threadIdx.x;
    float s = 0.f;
    for (int i = tid; i < 25 * HDIM;       i += 256) s += fabsf(Wi0[i]);
    for (int i = tid; i < HDIM;            i += 256) s += fabsf(bi0[i]);
    for (int i = tid; i < 3 * HDIM * HDIM; i += 256) s += fabsf(Wi[i]);
    for (int i = tid; i < 3 * HDIM;        i += 256) s += fabsf(bi[i]);
    red[tid] = s;
    __syncthreads();
    for (int o = 128; o > 0; o >>= 1) {
        if (tid < o) red[tid] += red[tid + o];
        __syncthreads();
    }
    if (tid == 0) out[0] = red[0];
}

// ---------------------------------------------------------------------------
extern "C" void kernel_launch(void* const* d_in, const int* in_sizes, int n_in,
                              void* d_out_v, int out_size, void* d_ws, size_t ws_size,
                              hipStream_t stream) {
    const float* x        = (const float*)d_in[0];
    const float* s        = (const float*)d_in[1];
    const int*   esrc     = (const int*)d_in[2];
    const int*   edst     = (const int*)d_in[3];
    const unsigned char* emask = (const unsigned char*)d_in[4];  // jnp bool -> 1 byte
    const float* W_self0  = (const float*)d_in[5];
    const float* b_self0  = (const float*)d_in[6];
    const float* W_intra0 = (const float*)d_in[7];
    const float* b_intra0 = (const float*)d_in[8];
    const float* W_inter0 = (const float*)d_in[9];
    const float* b_inter0 = (const float*)d_in[10];
    const float* W_self   = (const float*)d_in[11];
    const float* b_self   = (const float*)d_in[12];
    const float* W_intra  = (const float*)d_in[13];
    const float* b_intra  = (const float*)d_in[14];
    const float* W_inter  = (const float*)d_in[15];
    const float* b_inter  = (const float*)d_in[16];
    const float* ln_gamma = (const float*)d_in[17];
    const float* ln_beta  = (const float*)d_in[18];
    const float* fn_gamma = (const float*)d_in[19];
    const float* fn_beta  = (const float*)d_in[20];
    float* out = (float*)d_out_v;

    // d_out layout (flat f32, reference return order)
    float* out_frag = out;                          // [512,16,256]
    float* out_mask = out + 2097152;                // [512,16]
    float* out_node = out + 2105344;                // [32768,256]
    float* out_reg  = out + 10493952;               // scalar

    // Workspace carve-up
    char* ws = (char*)d_ws;
    size_t off = 0;
    auto alloc = [&](size_t bytes) -> char* {
        char* p = ws + off;
        off += (bytes + 255) & ~(size_t)255;
        return p;
    };
    bf16*  A0  = (bf16*)alloc((size_t)N_NODES * 32 * sizeof(bf16));
    bf16*  A   = (bf16*)alloc((size_t)N_NODES * HDIM * sizeof(bf16));
    float* Hs  = (float*)alloc((size_t)N_NODES * HDIM * sizeof(float));
    float* Ha  = (float*)alloc((size_t)N_NODES * HDIM * sizeof(float));
    float* Hr  = (float*)alloc((size_t)N_NODES * HDIM * sizeof(float));
    bf16*  Wp0 = (bf16*)alloc((size_t)3 * 32 * HDIM * sizeof(bf16));
    bf16*  WpL = (bf16*)alloc((size_t)9 * HDIM * HDIM * sizeof(bf16));
    int*   fid = (int*)alloc((size_t)N_NODES * sizeof(int));
    (void)ws_size; (void)n_in; (void)in_sizes; (void)out_size;

    // ---- stage inputs ----
    packx_kernel<<<(N_NODES * 32 + 255) / 256, 256, 0, stream>>>(x, A0, N_NODES);
    fragid_kernel<<<(N_NODES + 255) / 256, 256, 0, stream>>>(s, fid, N_NODES);

    const int kpack0 = 32 * HDIM;          // 8192 packed elems per layer-0 matrix
    const int kpackL = HDIM * HDIM;        // 65536 per layer-1..3 matrix
    packw_kernel<<<(kpack0 + 255) / 256, 256, 0, stream>>>(W_self0,  Wp0,              32, 25);
    packw_kernel<<<(kpack0 + 255) / 256, 256, 0, stream>>>(W_intra0, Wp0 + kpack0,     32, 25);
    packw_kernel<<<(kpack0 + 255) / 256, 256, 0, stream>>>(W_inter0, Wp0 + 2 * kpack0, 32, 25);
    for (int j = 0; j < 3; ++j) {
        packw_kernel<<<(kpackL + 255) / 256, 256, 0, stream>>>(W_self  + (size_t)j * kpackL, WpL + (size_t)(j * 3 + 0) * kpackL, HDIM, HDIM);
        packw_kernel<<<(kpackL + 255) / 256, 256, 0, stream>>>(W_intra + (size_t)j * kpackL, WpL + (size_t)(j * 3 + 1) * kpackL, HDIM, HDIM);
        packw_kernel<<<(kpackL + 255) / 256, 256, 0, stream>>>(W_inter + (size_t)j * kpackL, WpL + (size_t)(j * 3 + 2) * kpackL, HDIM, HDIM);
    }

    // ---- 4 conv layers ----
    dim3 ggrid(N_NODES / 128, HDIM / 64);
    for (int layer = 0; layer < 4; ++layer) {
        const bf16* Ain; int K;
        const bf16 *B0, *B1, *B2;
        const float *bs, *ba, *br;
        if (layer == 0) {
            Ain = A0; K = 32;
            B0 = Wp0; B1 = Wp0 + kpack0; B2 = Wp0 + 2 * kpack0;
            bs = b_self0; ba = b_intra0; br = b_inter0;
        } else {
            int j = layer - 1;
            Ain = A; K = HDIM;
            B0 = WpL + (size_t)(j * 3 + 0) * kpackL;
            B1 = WpL + (size_t)(j * 3 + 1) * kpackL;
            B2 = WpL + (size_t)(j * 3 + 2) * kpackL;
            bs = b_self  + j * HDIM;
            ba = b_intra + j * HDIM;
            br = b_inter + j * HDIM;
        }
        gemm3_kernel<<<ggrid, 256, 0, stream>>>(Ain, K, B0, B1, B2, bs, ba, br, Hs, Ha, Hr);
        fuse_kernel<<<NGRAPH, 256, 0, stream>>>(
            Hs, Ha, Hr, esrc, edst, emask,
            ln_gamma + layer * HDIM, ln_beta + layer * HDIM,
            (layer < 3) ? A : (bf16*)nullptr,
            (layer == 3) ? out_node : (float*)nullptr);
    }

    // ---- fragment pooling + mask ----
    frag_kernel<<<NGRAPH, 256, 0, stream>>>(out_node, fid, fn_gamma, fn_beta, out_frag, out_mask);

    // ---- reg loss ----
    regloss_kernel<<<1, 256, 0, stream>>>(W_inter0, b_inter0, W_inter, b_inter, out_reg);
}
